// GraphAttentionLayer_69638599737443
// MI455X (gfx1250) — matrix operand order repaired and checked by reading, not compiled, not run on hardware
//
#include <hip/hip_runtime.h>
#include <hip/hip_bf16.h>

typedef __attribute__((ext_vector_type(16))) __bf16 v16bf;
typedef __attribute__((ext_vector_type(8)))  __bf16 v8bf;
typedef __attribute__((ext_vector_type(4)))  __bf16 v4bf;
typedef __attribute__((ext_vector_type(8)))  float  v8f;
typedef __attribute__((ext_vector_type(4)))  float  v4f;
typedef __attribute__((ext_vector_type(4)))  int    v4i;

#define B_    8
#define N_    1600
#define V_    256
#define F_    256
#define ALPHA 0.2f

// ---------------------------------------------------------------------------
// Kernel 0: f32 -> bf16 conversion
// ---------------------------------------------------------------------------
__global__ void cvt_bf16(const float* __restrict__ src, __bf16* __restrict__ dst, int n) {
    int i = blockIdx.x * blockDim.x + threadIdx.x;
    if (i < n) dst[i] = (__bf16)src[i];
}

// ---------------------------------------------------------------------------
// Kernel 1: Wh = x * W^T + bW   via v_wmma_f32_16x16x32_bf16
//   x  : [B*N, V] bf16 (row major)      -> A fragments
//   W  : [F, V]   bf16 (row major)      -> B fragments (K contiguous per lane)
//   Wh : [B*N, F] f32
// One wave handles a 16-row m-tile x 64-col f-chunk. 3200 waves total.
// ---------------------------------------------------------------------------
__global__ void __launch_bounds__(256) wh_gemm(const __bf16* __restrict__ hb,
                                               const __bf16* __restrict__ Wb,
                                               const float*  __restrict__ bW,
                                               float*        __restrict__ Wh) {
    int wave = blockIdx.x * 8 + (threadIdx.x >> 5);   // 0..3199
    int lane = threadIdx.x & 31;
    int mt = wave >> 2;                               // 0..799  (16-row tiles of B*N)
    int fc = wave & 3;                                // 0..3    (64-col chunks of F)
    int li = lane & 15;
    int hiK = lane >> 4;

    const __bf16* aBase = hb + (size_t)(mt * 16 + li) * V_;

    v8f acc[4] = {};
    for (int k0 = 0; k0 < V_; k0 += 32) {
        // A fragment: row = li; elems 0..15 -> K=k0+hiK*16+c (contiguous)
        v8bf lo = *(const v8bf*)(aBase + k0 + hiK * 16);
        v8bf hh = *(const v8bf*)(aBase + k0 + 8 + hiK * 16);
        v16bf a;
#pragma unroll
        for (int c = 0; c < 8; ++c) { a[c] = lo[c]; a[c + 8] = hh[c]; }

#pragma unroll
        for (int t = 0; t < 4; ++t) {
            int f = fc * 64 + t * 16 + li;   // B column = lane&15
            // B fragment: K elems contiguous: K = k0 + hiK*16 + i
            v16bf bfrag = *(const v16bf*)(Wb + (size_t)f * V_ + k0 + hiK * 16);
            acc[t] = __builtin_amdgcn_wmma_f32_16x16x32_bf16(
                false, a, false, bfrag, (short)0, acc[t], false, false);
        }
    }

    // D layout: col = lane&15, rows = r + hiK*8
#pragma unroll
    for (int t = 0; t < 4; ++t) {
        int f = fc * 64 + t * 16 + li;
        float bias = bW[f];
#pragma unroll
        for (int r = 0; r < 8; ++r) {
            int row = mt * 16 + r + hiK * 8;
            Wh[(size_t)row * F_ + f] = acc[t][r] + bias;
        }
    }
}

// ---------------------------------------------------------------------------
// Kernel 2: per-(b,n) row:  s1 = Wh.ai + bi, s2 = Wh.aj + bj, WhT bf16 = Wh^T
// ---------------------------------------------------------------------------
__global__ void rows_kernel(const float* __restrict__ Wh,
                            const float* __restrict__ ai, const float* __restrict__ bi,
                            const float* __restrict__ aj, const float* __restrict__ bj,
                            float* __restrict__ s1, float* __restrict__ s2,
                            __bf16* __restrict__ WhT) {
    int idx = blockIdx.x * blockDim.x + threadIdx.x;    // b*N + n
    if (idx >= B_ * N_) return;
    int b = idx / N_, n = idx % N_;
    const float* row = Wh + (size_t)idx * F_;
    float d1 = 0.f, d2 = 0.f;
    for (int f = 0; f < F_; ++f) {
        float w = row[f];
        d1 += w * ai[f];
        d2 += w * aj[f];
        WhT[((size_t)b * F_ + f) * N_ + n] = (__bf16)w;
    }
    s1[idx] = d1 + bi[0];
    s2[idx] = d2 + bj[0];
}

// ---------------------------------------------------------------------------
// Kernel 3: masked-softmax stats per (b,i): row max, then P = exp(e-m) stored
// as bf16 [B][N][N] row-major (computed exactly once), plus Sum(exp).
// One wave per row; wave32 shuffle reductions.
// ---------------------------------------------------------------------------
__global__ void __launch_bounds__(256) stats_kernel(const int*   __restrict__ adj,
                                                    const float* __restrict__ s1,
                                                    const float* __restrict__ s2,
                                                    float* __restrict__ rowsum,
                                                    __bf16* __restrict__ P) {
    int wave = blockIdx.x * 8 + (threadIdx.x >> 5);   // 0..12799 == b*N + i
    int lane = threadIdx.x & 31;
    int b = wave / N_, i = wave % N_;
    float s1i = s1[wave];
    const float* s2b = s2 + b * N_;
    const int* arow = adj + (size_t)i * N_;

    // pass 1: row max over unmasked entries
    float m = -3.402823e38f;
    for (int j = lane; j < N_; j += 32) {
        if (arow[j] > 0) {
            float e = s1i + s2b[j];
            e = e > 0.f ? e : ALPHA * e;
            m = fmaxf(m, e);
        }
    }
#pragma unroll
    for (int o = 16; o >= 1; o >>= 1) m = fmaxf(m, __shfl_xor(m, o, 32));

    // pass 2: P[j] = adj ? exp(e - m) : 0  (bf16), and sum
    __bf16* prow = P + (size_t)wave * N_;
    float s = 0.f;
    for (int j0 = lane * 4; j0 < N_; j0 += 128) {
        v4i av = *(const v4i*)(arow + j0);
        v4f sv = *(const v4f*)(s2b + j0);
        v4bf pk;
#pragma unroll
        for (int c = 0; c < 4; ++c) {
            float e = s1i + sv[c];
            e = e > 0.f ? e : ALPHA * e;
            float p = (av[c] > 0) ? __expf(e - m) : 0.f;
            s += p;
            pk[c] = (__bf16)p;
        }
        *(v4bf*)(prow + j0) = pk;
    }
#pragma unroll
    for (int o = 16; o >= 1; o >>= 1) s += __shfl_xor(s, o, 32);

    if (lane == 0) rowsum[wave] = s;
}

// ---------------------------------------------------------------------------
// Kernel 4: h_hat = (P * Wh) / rowsum, then ELU. Pure streaming bf16 WMMA GEMM:
//   A = P [B*N, N] bf16 row-major, B = WhT [B][F][N] bf16 (K contiguous).
// One wave: 16 i-rows x 64 f-cols, K loop over 1600 in steps of 32.
// ---------------------------------------------------------------------------
__global__ void __launch_bounds__(256) attn_gemm(const __bf16* __restrict__ P,
                                                 const float*  __restrict__ rowsum,
                                                 const __bf16* __restrict__ WhT,
                                                 float*        __restrict__ out) {
    int wave = blockIdx.x * 8 + (threadIdx.x >> 5);   // 0..3199
    int lane = threadIdx.x & 31;
    int b   = wave / 400;
    int rem = wave % 400;
    int it  = rem >> 2;          // 0..99  (16-row i tile)
    int fc  = rem & 3;           // 0..3   (64-col f chunk)
    int li  = lane & 15;
    int hi  = lane >> 4;

    const __bf16* aBase = P + (size_t)(b * N_ + it * 16 + li) * N_;

    v8f acc[4] = {};
    for (int j0 = 0; j0 < N_; j0 += 32) {
        if (j0 + 32 < N_)   // gfx1250 global_prefetch_b8 of next A line
            __builtin_prefetch((const void*)(aBase + j0 + 32 + hi * 8), 0, 3);
        // A fragment: elems 0..15 -> K=j0+hi*16+c (contiguous)
        v8bf lo = *(const v8bf*)(aBase + j0 + hi * 16);
        v8bf hh = *(const v8bf*)(aBase + j0 + 8 + hi * 16);
        v16bf a;
#pragma unroll
        for (int c = 0; c < 8; ++c) { a[c] = lo[c]; a[c + 8] = hh[c]; }

#pragma unroll
        for (int t = 0; t < 4; ++t) {
            int f = fc * 64 + t * 16 + li;
            v16bf bfrag = *(const v16bf*)(WhT + ((size_t)b * F_ + f) * N_ + j0 + hi * 16);
            acc[t] = __builtin_amdgcn_wmma_f32_16x16x32_bf16(
                false, a, false, bfrag, (short)0, acc[t], false, false);
        }
    }

    float inv[8];
#pragma unroll
    for (int r = 0; r < 8; ++r) {
        int row = it * 16 + r + hi * 8;
        inv[r] = 1.f / rowsum[b * N_ + row];
    }
#pragma unroll
    for (int t = 0; t < 4; ++t) {
        int f = fc * 64 + t * 16 + li;
#pragma unroll
        for (int r = 0; r < 8; ++r) {
            int row = it * 16 + r + hi * 8;
            float v = acc[t][r] * inv[r];
            v = v > 0.f ? v : (__expf(v) - 1.f);     // ELU
            out[((size_t)(b * N_ + row)) * F_ + f] = v;
        }
    }
}

// ---------------------------------------------------------------------------
extern "C" void kernel_launch(void* const* d_in, const int* in_sizes, int n_in,
                              void* d_out, int out_size, void* d_ws, size_t ws_size,
                              hipStream_t stream) {
    (void)in_sizes; (void)n_in; (void)out_size; (void)ws_size;
    const float* h   = (const float*)d_in[0];
    const int*   adj = (const int*)d_in[1];
    const float* W   = (const float*)d_in[2];
    const float* bW  = (const float*)d_in[3];
    const float* ai  = (const float*)d_in[4];
    const float* bi  = (const float*)d_in[5];
    const float* aj  = (const float*)d_in[6];
    const float* bj  = (const float*)d_in[7];
    float* out = (float*)d_out;

    char* ws = (char*)d_ws;
    size_t off = 0;
    auto alloc = [&](size_t bytes) -> void* {
        void* p = ws + off;
        off = (off + bytes + 255) & ~(size_t)255;
        return p;
    };
    __bf16* hb  = (__bf16*)alloc((size_t)B_ * N_ * V_ * 2);   // 6.55 MB
    __bf16* Wb  = (__bf16*)alloc((size_t)F_ * V_ * 2);        // 131 KB
    float*  Wh  = (float*) alloc((size_t)B_ * N_ * F_ * 4);   // 13.1 MB
    __bf16* WhT = (__bf16*)alloc((size_t)B_ * F_ * N_ * 2);   // 6.55 MB
    __bf16* P   = (__bf16*)alloc((size_t)B_ * N_ * N_ * 2);   // 40.96 MB
    float*  s1v = (float*) alloc((size_t)B_ * N_ * 4);
    float*  s2v = (float*) alloc((size_t)B_ * N_ * 4);
    float*  rs  = (float*) alloc((size_t)B_ * N_ * 4);

    cvt_bf16<<<(B_ * N_ * V_ + 255) / 256, 256, 0, stream>>>(h, hb, B_ * N_ * V_);
    cvt_bf16<<<(F_ * V_ + 255) / 256, 256, 0, stream>>>(W, Wb, F_ * V_);
    wh_gemm<<<400, 256, 0, stream>>>(hb, Wb, bW, Wh);                    // 3200 waves
    rows_kernel<<<50, 256, 0, stream>>>(Wh, ai, bi, aj, bj, s1v, s2v, WhT);
    stats_kernel<<<1600, 256, 0, stream>>>(adj, s1v, s2v, rs, P);        // 12800 waves
    attn_gemm<<<400, 256, 0, stream>>>(P, rs, WhT, out);                 // 3200 waves
}
